// GatedLRU_695784702453
// MI455X (gfx1250) — matrix operand order repaired and checked by reading, compiled, not run on hardware
//
#include <hip/hip_runtime.h>
#include <hip/hip_bf16.h>
#include <math.h>

typedef _Float16 f16_t;
typedef __attribute__((ext_vector_type(16))) _Float16 v16h;
typedef __attribute__((ext_vector_type(8)))  _Float16 v8h;
typedef __attribute__((ext_vector_type(8)))  float    v8f;
typedef __attribute__((ext_vector_type(4)))  float    v4f;

#define D_MODEL 1024
#define D_STATE 64
#define CHUNK   64
#define NBATCH  4
#define SEQ     4096
#define NCHUNK  (SEQ / CHUNK)          // 64
#define NROWS   (NBATCH * SEQ)         // 16384
#define NBC     (NBATCH * NCHUNK)      // 256 chunk-blocks

#define WROW    72                      // LDS row stride (halves) for W slice

__device__ __forceinline__ float sigf(float x) { return 1.0f / (1.0f + __expf(-x)); }

// Async copy one 16B chunk into LDS (CDNA5, ASYNCcnt-tracked).
__device__ __forceinline__ void async_b128_to_lds(const f16_t* gsrc, f16_t* ldst) {
    unsigned lds_off = (unsigned)(uintptr_t)ldst;                 // LDS byte offset
    unsigned long long ga = (unsigned long long)(uintptr_t)gsrc;  // 64-bit global addr
    asm volatile("global_load_async_to_lds_b128 %0, %1, off"
                 :: "v"(lds_off), "v"(ga) : "memory");
}

__device__ __forceinline__ void wait_async_all() {
    asm volatile("s_wait_asynccnt 0x0" ::: "memory");
}

// Stage W_in_h[k-slice]: all 128 rows (n), 32 halves starting at k0.
// LDS layout: row n at dst + n*WROW halves (64B payload per row).
// 128 threads x 4 issues = 512 x 16B = 8KB.
__device__ __forceinline__ void stage_w_slice(f16_t* dst, const f16_t* W, int k0, int tid) {
    #pragma unroll
    for (int i = 0; i < 4; ++i) {
        const int c = tid + 128 * i;       // 0..511
        const int n = c >> 2;              // row
        const int p = c & 3;               // 16B part within 64B row payload
        async_b128_to_lds(W + (size_t)n * D_MODEL + k0 + p * 8,
                          dst + (size_t)n * WROW + p * 8);
    }
}

// ---------------------------------------------------------------------------
// Prep: convert W_in [128,1024] and W_out [1024,64] to f16; build a[d] and
// apow[j][d] = a[d]^(j+1).
// ---------------------------------------------------------------------------
__global__ void k_prep(const float* __restrict__ W_in, const float* __restrict__ W_out,
                       const float* __restrict__ rec_w, const float* __restrict__ pos_bias,
                       f16_t* __restrict__ W_in_h, f16_t* __restrict__ W_out_h,
                       float* __restrict__ a_vec, float* __restrict__ apow) {
    int i = blockIdx.x * 256 + threadIdx.x;
    if (i < 2 * D_STATE * D_MODEL) W_in_h[i] = (f16_t)W_in[i];
    if (i < D_MODEL * D_STATE)     W_out_h[i] = (f16_t)W_out[i];
    if (i < D_STATE)               a_vec[i] = sigf(rec_w[i] + pos_bias[i]);
    if (i < CHUNK * D_STATE) {
        int j = i >> 6, d = i & 63;
        float ad = sigf(rec_w[d] + pos_bias[d]);
        apow[i] = __powf(ad, (float)(j + 1));
    }
}

// ---------------------------------------------------------------------------
// Kernel 1: proj = x@W_in^T + b_in  (f16 WMMA, f32 accum), gating to u in LDS,
// then the within-chunk local scan. W_in K-slices are double-buffered into LDS
// via global_load_async_to_lds_b128 so B-fragment traffic is block-shared and
// overlapped with WMMA compute.
// Block = one chunk (64 rows) x 128 proj cols, 128 threads = 4 waves.
// ---------------------------------------------------------------------------
__global__ void __launch_bounds__(128) k_proj_gate_scan(
    const float* __restrict__ x, const f16_t* __restrict__ W_in_h,
    const float* __restrict__ b_in, const float* __restrict__ a_vec,
    float* __restrict__ s_local, float* __restrict__ carries) {
    __shared__ float u_lds[CHUNK * 65];
    __shared__ __align__(16) f16_t Wlds[2][128 * WROW];   // 2 x 18KB

    const int bc    = blockIdx.x;       // b*NCHUNK + chunk; rows are contiguous
    const int r0    = bc * CHUNK;       // first global row of this chunk
    const int tid   = threadIdx.x;
    const int wave  = tid >> 5;
    const int lane  = tid & 31;
    const int lhalf = lane >> 4;        // 0: lanes 0-15, 1: lanes 16-31
    const int l16   = lane & 15;

    const size_t m = (size_t)(r0 + wave * 16 + l16);   // A-matrix row (lane-local)

    // prime slice 0
    stage_w_slice(Wlds[0], W_in_h, 0, tid);
    wait_async_all();
    __syncthreads();

    v8f acc[8] = {};

    for (int k0 = 0, it = 0; k0 < D_MODEL; k0 += 32, ++it) {
        const f16_t* Wc = Wlds[it & 1];
        const bool more = (k0 + 32) < D_MODEL;
        if (more) stage_w_slice(Wlds[(it + 1) & 1], W_in_h, k0 + 32, tid);

        // ---- A fragment: row m, K = {kb..kb+7, kb+16..kb+23}, kb = k0 + lhalf*8
        const float* xp = x + m * D_MODEL + k0 + lhalf * 8;
        __builtin_prefetch(xp + 256, 0, 3);            // global_prefetch_b8
        v4f a0 = *(const v4f*)(xp);
        v4f a1 = *(const v4f*)(xp + 4);
        v4f a2 = *(const v4f*)(xp + 16);
        v4f a3 = *(const v4f*)(xp + 20);
        v16h af;
        #pragma unroll
        for (int i = 0; i < 4; ++i) {
            af[i]      = (f16_t)a0[i];
            af[4 + i]  = (f16_t)a1[i];
            af[8 + i]  = (f16_t)a2[i];
            af[12 + i] = (f16_t)a3[i];
        }
        // ---- 8 B fragments from LDS: lane: col n = nt*16+l16, K = {kb..kb+7, kb+16..kb+23}, kb = lhalf*8
        #pragma unroll
        for (int nt = 0; nt < 8; ++nt) {
            const f16_t* wp = Wc + (size_t)(nt * 16 + l16) * WROW + lhalf * 8;
            v8h blo = *(const v8h*)(wp);
            v8h bhi = *(const v8h*)(wp + 16);
            v16h bf = __builtin_shufflevector(blo, bhi,
                        0,1,2,3,4,5,6,7,8,9,10,11,12,13,14,15);
            acc[nt] = __builtin_amdgcn_wmma_f32_16x16x32_f16(
                false, af, false, bf, (short)0, acc[nt], false, false);
        }

        if (more) {            // publish next slice to all waves
            wait_async_all();
            __syncthreads();
        }
    }

    // ---- gating: u = sigmoid(proj[d+64]) * tanh(proj[d]); stash in LDS
    #pragma unroll
    for (int nt = 0; nt < 4; ++nt) {
        const int d  = nt * 16 + l16;
        const float bi = b_in[d];
        const float bg = b_in[D_STATE + d];
        #pragma unroll
        for (int r = 0; r < 8; ++r) {
            const int j  = wave * 16 + r + 8 * lhalf;   // row within chunk
            float pv = acc[nt][r] + bi;
            float pg = acc[nt + 4][r] + bg;
            u_lds[j * 65 + d] = sigf(pg) * tanhf(pv);
        }
    }
    __syncthreads();

    // ---- within-chunk scan, one thread per state dim
    if (tid < D_STATE) {
        const int d = tid;
        const float ad = a_vec[d];
        float h = 0.0f;
        #pragma unroll 4
        for (int j = 0; j < CHUNK; ++j) {
            h = fmaf(ad, h, u_lds[j * 65 + d]);
            s_local[(size_t)(r0 + j) * D_STATE + d] = h;
        }
        carries[bc * D_STATE + d] = h;
    }
}

// ---------------------------------------------------------------------------
// Kernel 2: scan chunk carries: h_in[c] = h entering chunk c,
//   h <- carries[c] + h * a^CHUNK.  Also emits h_last to tail of d_out.
// One block of 256 threads: thread = b*64 + d.
// ---------------------------------------------------------------------------
__global__ void k_carry_scan(const float* __restrict__ state0,
                             const float* __restrict__ apow,
                             const float* __restrict__ carries,
                             float* __restrict__ h_in,
                             float* __restrict__ h_last_out) {
    const int t = threadIdx.x;
    const int b = t >> 6, d = t & 63;
    const float aC = apow[(CHUNK - 1) * D_STATE + d];   // a^64
    float h = state0[b * D_STATE + d];
    for (int c = 0; c < NCHUNK; ++c) {
        const int idx = (b * NCHUNK + c) * D_STATE + d;
        h_in[idx] = h;
        h = fmaf(h, aC, carries[idx]);
    }
    h_last_out[b * D_STATE + d] = h;
}

// ---------------------------------------------------------------------------
// Kernel 3: states[j][d] = s_local + h_in * a^(j+1)  (staged to LDS as f16),
// then out = states @ W_out^T + b_out via f16 WMMA (K=64 -> 2 WMMAs / tile).
// Block = one chunk (64 rows) x 1024 out cols, 256 threads = 8 waves.
// Wave (rw = w&3, cw = w>>2): rows 16*rw..+15, cols cw*512 .. +511.
// ---------------------------------------------------------------------------
__global__ void __launch_bounds__(256) k_fix_outproj(
    const float* __restrict__ s_local, const float* __restrict__ h_in,
    const float* __restrict__ apow, const f16_t* __restrict__ W_out_h,
    const float* __restrict__ b_out, float* __restrict__ out) {
    __shared__ __align__(32) f16_t st[CHUNK * 72];

    const int bc  = blockIdx.x;
    const int r0  = bc * CHUNK;
    const int tid = threadIdx.x;

    // build corrected states tile in LDS (f16)
    for (int e = tid; e < CHUNK * D_STATE; e += 256) {
        const int j = e >> 6, d = e & 63;
        float v = s_local[(size_t)(r0 + j) * D_STATE + d]
                + h_in[bc * D_STATE + d] * apow[j * D_STATE + d];
        st[j * 72 + d] = (f16_t)v;
    }
    __syncthreads();

    const int wave  = tid >> 5;
    const int lane  = tid & 31;
    const int rw    = wave & 3;
    const int cw    = wave >> 2;
    const int lhalf = lane >> 4;
    const int l16   = lane & 15;

    // A fragments for K=0..31 and K=32..63
    const f16_t* arow = &st[(rw * 16 + l16) * 72];
    v8h a0lo = *(const v8h*)(arow + lhalf * 16);
    v8h a0hi = *(const v8h*)(arow + 8 + lhalf * 16);
    v8h a1lo = *(const v8h*)(arow + 32 + lhalf * 16);
    v8h a1hi = *(const v8h*)(arow + 40 + lhalf * 16);
    v16h af0 = __builtin_shufflevector(a0lo, a0hi, 0,1,2,3,4,5,6,7,8,9,10,11,12,13,14,15);
    v16h af1 = __builtin_shufflevector(a1lo, a1hi, 0,1,2,3,4,5,6,7,8,9,10,11,12,13,14,15);

    for (int nt = 0; nt < 32; ++nt) {
        const int n = cw * 512 + nt * 16 + l16;
        const f16_t* wp = W_out_h + (size_t)n * D_STATE + lhalf * 16;
        v16h bf0 = *(const v16h*)(wp);
        v16h bf1 = *(const v16h*)(wp + 32);
        v8f acc = {};
        acc = __builtin_amdgcn_wmma_f32_16x16x32_f16(false, af0, false, bf0, (short)0, acc, false, false);
        acc = __builtin_amdgcn_wmma_f32_16x16x32_f16(false, af1, false, bf1, (short)0, acc, false, false);
        const float bo = b_out[n];
        #pragma unroll
        for (int r = 0; r < 8; ++r) {
            const int row = r0 + rw * 16 + r + 8 * lhalf;
            out[(size_t)row * D_MODEL + n] = acc[r] + bo;
        }
    }
}

// ---------------------------------------------------------------------------
extern "C" void kernel_launch(void* const* d_in, const int* in_sizes, int n_in,
                              void* d_out, int out_size, void* d_ws, size_t ws_size,
                              hipStream_t stream) {
    const float* x        = (const float*)d_in[0];
    const float* state0   = (const float*)d_in[1];
    const float* W_in     = (const float*)d_in[2];
    const float* b_in     = (const float*)d_in[3];
    const float* rec_w    = (const float*)d_in[4];
    const float* pos_bias = (const float*)d_in[5];
    const float* W_out    = (const float*)d_in[6];
    const float* b_out    = (const float*)d_in[7];

    float* out    = (float*)d_out;                          // [B,T,D_MODEL]
    float* h_last = out + (size_t)NROWS * D_MODEL;          // [B,D_STATE] tail

    // workspace carve-up
    char* ws = (char*)d_ws;
    f16_t* W_in_h  = (f16_t*)ws;                 ws += (size_t)2 * D_STATE * D_MODEL * 2; // 256 KB
    f16_t* W_out_h = (f16_t*)ws;                 ws += (size_t)D_MODEL * D_STATE * 2;     // 128 KB
    float* s_local = (float*)ws;                 ws += (size_t)NROWS * D_STATE * 4;       // 4 MB
    float* carries = (float*)ws;                 ws += (size_t)NBC * D_STATE * 4;         // 64 KB
    float* h_in    = (float*)ws;                 ws += (size_t)NBC * D_STATE * 4;         // 64 KB
    float* a_vec   = (float*)ws;                 ws += (size_t)D_STATE * 4;
    float* apow    = (float*)ws;                 ws += (size_t)CHUNK * D_STATE * 4;       // 16 KB

    k_prep<<<512, 256, 0, stream>>>(W_in, W_out, rec_w, pos_bias,
                                    W_in_h, W_out_h, a_vec, apow);
    k_proj_gate_scan<<<NBC, 128, 0, stream>>>(x, W_in_h, b_in, a_vec,
                                              s_local, carries);
    k_carry_scan<<<1, 256, 0, stream>>>(state0, apow, carries, h_in, h_last);
    k_fix_outproj<<<NBC, 256, 0, stream>>>(s_local, h_in, apow, W_out_h,
                                           b_out, out);
}
